// GINNet_1726576853642
// MI455X (gfx1250) — matrix operand order repaired and checked by reading, not compiled, not run on hardware
//
#include <hip/hip_runtime.h>

// ---------------------------------------------------------------------------
// GIN network for MI455X (gfx1250): bf16 WMMA fused MLPs + L2-resident scatter
// ---------------------------------------------------------------------------

#define N_NODES  40000
#define N_EDGES  640000
#define N_FEAT   64
#define HIDDEN   128
#define N_GRAPHS 256

typedef __attribute__((ext_vector_type(16))) __bf16 v16bf;
typedef __attribute__((ext_vector_type(8)))  __bf16 v8bf;
typedef __attribute__((ext_vector_type(8)))  float  v8f;

// ---------------------------------------------------------------------------
// Utility: zero an f32 buffer
// ---------------------------------------------------------------------------
__global__ void zero_kernel(float* __restrict__ p, int n) {
    int i = blockIdx.x * 256 + threadIdx.x;
    if (i < n) p[i] = 0.0f;
}

// ---------------------------------------------------------------------------
// Pack a KxN(=128) f32 weight matrix into WMMA bf16 B-fragment order:
//   frag[((n_tile*ksteps + kstep)*512) + lane*16 + pos]
// lane = (N%16) + 16*((K%32)>=16), pos = K%16   (contiguous k order as in load_a_frag, K=32)
// Each lane's 16 halfwords become one contiguous 32B load in the GEMM.
// ---------------------------------------------------------------------------
__global__ void pack_weight_kernel(const float* __restrict__ W,
                                   __bf16* __restrict__ out, int K) {
    int idx = blockIdx.x * 256 + threadIdx.x;
    if (idx >= K * 128) return;
    int k = idx >> 7;          // row (K dim)
    int n = idx & 127;         // col (N dim)
    int ksteps = K >> 5;
    int lane = (n & 15) | (((k >> 4) & 1) << 4);
    int pos  = k & 15;
    int off  = (((n >> 4) * ksteps + (k >> 5)) * 512) + lane * 16 + pos;
    out[off] = (__bf16)W[idx];
}

// ---------------------------------------------------------------------------
// Scatter-add: agg[dst] += h[src]  (per edge, F features). L2-resident atomics.
// ---------------------------------------------------------------------------
template <int F>
__global__ void scatter_add_kernel(const float* __restrict__ h,
                                   const int* __restrict__ ei,
                                   float* __restrict__ agg) {
    const int epb = 256 / F;
    int e = blockIdx.x * epb + (threadIdx.x / F);
    int f = threadIdx.x & (F - 1);
    if (e < N_EDGES) {
        int s = ei[e];
        int d = ei[N_EDGES + e];
        atomicAdd(&agg[d * F + f], h[s * F + f]);
    }
}

// ---------------------------------------------------------------------------
// A-fragment load from an LDS tile stored row-major [16][128] bf16.
// A layout matching packed B: lane L holds row M=L%16;
//   lanes 0-15 : K = kb+{0..15}; lanes 16-31: K = kb+{16..31}
// -> one contiguous 32B chunk per lane.
// ---------------------------------------------------------------------------
__device__ inline v16bf load_a_frag(const __bf16* z, int lane, int kb) {
    int m  = lane & 15;
    int hi = lane >> 4;
    v16bf a = *(const v16bf*)(z + m * 128 + kb + hi * 16);
    return a;
}

// ---------------------------------------------------------------------------
// Fused GIN MLP: out = relu( relu( ((1+eps)h + agg) @ Wa + ba ) @ Wb + bb )
// One workgroup = 16 nodes (M tile) x 128 outputs; 8 waves, wave w owns
// output columns [16w, 16w+16). Intermediate activation stays in LDS (bf16).
// ---------------------------------------------------------------------------
template <int K1>
__global__ void __launch_bounds__(256)
gin_mlp_kernel(const float* __restrict__ h_in, const float* __restrict__ agg,
               const float* __restrict__ eps_p,
               const __bf16* __restrict__ Wa, const float* __restrict__ ba,
               const __bf16* __restrict__ Wb, const float* __restrict__ bb,
               float* __restrict__ h_out) {
    __shared__ alignas(32) __bf16 zA[16 * 128];
    __shared__ alignas(32) __bf16 zB[16 * 128];

    const int tid  = threadIdx.x;
    const int lane = tid & 31;
    const int w    = tid >> 5;              // wave id = output n-tile
    const int base = blockIdx.x * 16;       // node tile (40000 = 16*2500 exact)
    const float ep = 1.0f + eps_p[0];

    // Hint the far cache about GEMM2 weights while we stage GEMM1.
    __builtin_prefetch(Wb + (w * 4) * 512 + lane * 16, 0, 1);

    // Stage z = (1+eps)*h + agg into LDS as bf16, row-major [16][128], first K1 cols
    constexpr int TOT = 16 * K1;
#pragma unroll
    for (int j = 0; j < TOT / 256; ++j) {
        int idx = tid + j * 256;
        int m = idx / K1, k = idx % K1;     // K1 is 64 or 128 -> shifts
        float v = ep * h_in[(base + m) * K1 + k] + agg[(base + m) * K1 + k];
        zA[m * 128 + k] = (__bf16)v;
    }
    __syncthreads();

    // ---- GEMM1: z[16xK1] @ Wa[K1x128], f32 accumulate ----
    v8f acc = {};
    constexpr int KS1 = K1 / 32;
#pragma unroll
    for (int ks = 0; ks < KS1; ++ks) {
        v16bf a = load_a_frag(zA, lane, ks * 32);
        v16bf b = *(const v16bf*)(Wa + (w * KS1 + ks) * 512 + lane * 16);
        acc = __builtin_amdgcn_wmma_f32_16x16x32_bf16(
            false, a, false, b, (short)0, acc, false, false);
    }
    {
        int n  = (w << 4) | (lane & 15);
        int hi = lane >> 4;
        float bias = ba[n];
#pragma unroll
        for (int r = 0; r < 8; ++r) {
            float v = fmaxf(acc[r] + bias, 0.0f);   // relu
            zB[(r + 8 * hi) * 128 + n] = (__bf16)v; // C/D layout: M = r + 8*hi
        }
    }
    __syncthreads();

    // ---- GEMM2: zB[16x128] @ Wb[128x128], f32 accumulate ----
    v8f acc2 = {};
#pragma unroll
    for (int ks = 0; ks < 4; ++ks) {
        v16bf a = load_a_frag(zB, lane, ks * 32);
        v16bf b = *(const v16bf*)(Wb + (w * 4 + ks) * 512 + lane * 16);
        acc2 = __builtin_amdgcn_wmma_f32_16x16x32_bf16(
            false, a, false, b, (short)0, acc2, false, false);
    }
    {
        int n  = (w << 4) | (lane & 15);
        int hi = lane >> 4;
        float bias = bb[n];
#pragma unroll
        for (int r = 0; r < 8; ++r) {
            float v = fmaxf(acc2[r] + bias, 0.0f);  // relu
            h_out[(base + r + 8 * hi) * 128 + n] = v;
        }
    }
}

// ---------------------------------------------------------------------------
// Global add pool: g[batch[v]] += h[v]
// ---------------------------------------------------------------------------
__global__ void pool_kernel(const float* __restrict__ h,
                            const int* __restrict__ batch,
                            float* __restrict__ g) {
    int idx = blockIdx.x * 256 + threadIdx.x;  // over N_NODES*128
    int node = idx >> 7, f = idx & 127;
    if (node < N_NODES) atomicAdd(&g[batch[node] * 128 + f], h[idx]);
}

// ---------------------------------------------------------------------------
// Head: out = relu(g @ W1 + b1) @ W2 + b2, one block per graph (tiny, f32)
// ---------------------------------------------------------------------------
__global__ void head_kernel(const float* __restrict__ g,
                            const float* __restrict__ W1,
                            const float* __restrict__ b1,
                            const float* __restrict__ W2,
                            const float* __restrict__ b2,
                            float* __restrict__ out) {
    __shared__ float s[128];
    int gi = blockIdx.x, t = threadIdx.x;
    float acc = b1[t];
    for (int k = 0; k < 128; ++k) acc += g[gi * 128 + k] * W1[k * 128 + t];
    acc = fmaxf(acc, 0.0f);
    s[t] = acc * W2[t];
    __syncthreads();
    for (int off = 64; off > 0; off >>= 1) {
        if (t < off) s[t] += s[t + off];
        __syncthreads();
    }
    if (t == 0) out[gi] = s[0] + b2[0];
}

// ---------------------------------------------------------------------------
// Launch
// ---------------------------------------------------------------------------
extern "C" void kernel_launch(void* const* d_in, const int* in_sizes, int n_in,
                              void* d_out, int out_size, void* d_ws,
                              size_t ws_size, hipStream_t stream) {
    (void)in_sizes; (void)n_in; (void)out_size; (void)ws_size;

    const float* x      = (const float*)d_in[0];
    const int*   ei     = (const int*)d_in[1];
    // d_in[2] = edge_attr (unused by reference math)
    const int*   batch  = (const int*)d_in[3];
    const float* eps1   = (const float*)d_in[4];
    const float* W1a    = (const float*)d_in[5];
    const float* b1a    = (const float*)d_in[6];
    const float* W1b    = (const float*)d_in[7];
    const float* b1b    = (const float*)d_in[8];
    const float* eps2   = (const float*)d_in[9];
    const float* W2a    = (const float*)d_in[10];
    const float* b2a    = (const float*)d_in[11];
    const float* W2b    = (const float*)d_in[12];
    const float* b2b    = (const float*)d_in[13];
    const float* eps3   = (const float*)d_in[14];
    const float* W3a    = (const float*)d_in[15];
    const float* b3a    = (const float*)d_in[16];
    const float* W3b    = (const float*)d_in[17];
    const float* b3b    = (const float*)d_in[18];
    const float* lin1_W = (const float*)d_in[19];
    const float* lin1_b = (const float*)d_in[20];
    const float* lin2_W = (const float*)d_in[21];
    const float* lin2_b = (const float*)d_in[22];
    float* out = (float*)d_out;

    // ---- workspace layout (all offsets 256B aligned) ----
    char* ws = (char*)d_ws;
    const size_t SZ_NODE = (size_t)N_NODES * 128 * sizeof(float);   // 20.48 MB
    float* agg = (float*)(ws);                      // reused at F=64 and F=128
    float* hA  = (float*)(ws + SZ_NODE);
    float* hB  = (float*)(ws + 2 * SZ_NODE);
    float* g   = (float*)(ws + 3 * SZ_NODE);        // 256*128 f32
    char*  wf  = ws + 3 * SZ_NODE + (size_t)N_GRAPHS * 128 * sizeof(float);
    __bf16* fW1a = (__bf16*)(wf);                               // 64*128
    __bf16* fW1b = (__bf16*)(wf + 16384);                       // 128*128
    __bf16* fW2a = (__bf16*)(wf + 16384 + 1 * 32768);
    __bf16* fW2b = (__bf16*)(wf + 16384 + 2 * 32768);
    __bf16* fW3a = (__bf16*)(wf + 16384 + 3 * 32768);
    __bf16* fW3b = (__bf16*)(wf + 16384 + 4 * 32768);

    // ---- pack weights into WMMA B-fragment order (bf16) ----
    pack_weight_kernel<<<(64 * 128 + 255) / 256, 256, 0, stream>>>(W1a, fW1a, 64);
    pack_weight_kernel<<<(128 * 128 + 255) / 256, 256, 0, stream>>>(W1b, fW1b, 128);
    pack_weight_kernel<<<(128 * 128 + 255) / 256, 256, 0, stream>>>(W2a, fW2a, 128);
    pack_weight_kernel<<<(128 * 128 + 255) / 256, 256, 0, stream>>>(W2b, fW2b, 128);
    pack_weight_kernel<<<(128 * 128 + 255) / 256, 256, 0, stream>>>(W3a, fW3a, 128);
    pack_weight_kernel<<<(128 * 128 + 255) / 256, 256, 0, stream>>>(W3b, fW3b, 128);

    const int GIN_BLOCKS = N_NODES / 16;  // 2500

    // ---- layer 1 (F = 64) ----
    zero_kernel<<<(N_NODES * 64 + 255) / 256, 256, 0, stream>>>(agg, N_NODES * 64);
    scatter_add_kernel<64><<<N_EDGES / 4, 256, 0, stream>>>(x, ei, agg);
    gin_mlp_kernel<64><<<GIN_BLOCKS, 256, 0, stream>>>(
        x, agg, eps1, fW1a, b1a, fW1b, b1b, hA);

    // ---- layer 2 (F = 128) ----
    zero_kernel<<<(N_NODES * 128 + 255) / 256, 256, 0, stream>>>(agg, N_NODES * 128);
    scatter_add_kernel<128><<<N_EDGES / 2, 256, 0, stream>>>(hA, ei, agg);
    gin_mlp_kernel<128><<<GIN_BLOCKS, 256, 0, stream>>>(
        hA, agg, eps2, fW2a, b2a, fW2b, b2b, hB);

    // ---- layer 3 (F = 128) ----
    zero_kernel<<<(N_NODES * 128 + 255) / 256, 256, 0, stream>>>(agg, N_NODES * 128);
    scatter_add_kernel<128><<<N_EDGES / 2, 256, 0, stream>>>(hB, ei, agg);
    gin_mlp_kernel<128><<<GIN_BLOCKS, 256, 0, stream>>>(
        hB, agg, eps3, fW3a, b3a, fW3b, b3b, hA);

    // ---- global add pool + head ----
    zero_kernel<<<(N_GRAPHS * 128 + 255) / 256, 256, 0, stream>>>(g, N_GRAPHS * 128);
    pool_kernel<<<(N_NODES * 128 + 255) / 256, 256, 0, stream>>>(hA, batch, g);
    head_kernel<<<N_GRAPHS, 128, 0, stream>>>(g, lin1_W, lin1_b, lin2_W, lin2_b, out);
}
